// RNN_24077586661463
// MI455X (gfx1250) — matrix operand order repaired and checked by reading, not compiled, not run on hardware
//
#include <hip/hip_runtime.h>

typedef _Float16 v8h  __attribute__((ext_vector_type(8)));
typedef _Float16 v16h __attribute__((ext_vector_type(16)));
typedef float    v8f  __attribute__((ext_vector_type(8)));

constexpr int Bsz = 64, Tsz = 256, Esz = 32, Hsz = 128, Vsz = 8000;

__device__ __forceinline__ v16h ld_frag(const _Float16* p0, const _Float16* p1) {
    v8h lo = *(const v8h*)p0;
    v8h hi = *(const v8h*)p1;
    return __builtin_shufflevector(lo, hi, 0,1,2,3,4,5,6,7,8,9,10,11,12,13,14,15);
}

__device__ __forceinline__ v8f wmma_f16(v16h a, v16h b, v8f c) {
    return __builtin_amdgcn_wmma_f32_16x16x32_f16(false, a, false, b, (short)0, c, false, false);
}

// Branch-free saturating tanh: e = exp(-2|x|) in (0,1] -> no overflow, no branches.
// tanh(x) = sign(x) * (1-e)/(1+e). Lowers to v_exp_f32 + v_rcp_f32 + a few VALU.
__device__ __forceinline__ float fast_tanh(float x) {
    float e = __expf(-2.0f * __builtin_fabsf(x));
    float r = __builtin_amdgcn_rcpf(1.0f + e);
    return __builtin_copysignf((1.0f - e) * r, x);
}

// Phase 1: ux[t][b][n] = x[b,t,:] . U_w[n,:] + U_b[n] + W_b[n]
__global__ __launch_bounds__(256) void ux_kernel(
    const float* __restrict__ x, const float* __restrict__ Uw,
    const float* __restrict__ Ub, const float* __restrict__ Wb,
    float* __restrict__ ux) {
    int id = blockIdx.x * 256 + threadIdx.x;          // (t*64 + b)*128 + n
    int n = id & (Hsz - 1);
    int rem = id >> 7;
    int b = rem & (Bsz - 1);
    int t = rem >> 6;
    const float* xr = x + ((size_t)b * Tsz + t) * Esz;
    const float* ur = Uw + n * Esz;
    float s = Ub[n] + Wb[n];
#pragma unroll
    for (int e = 0; e < Esz; ++e) s += xr[e] * ur[e];
    ux[id] = s;
}

// f32 -> f16 convert (for V_w)
__global__ __launch_bounds__(256) void cvt_kernel(const float* __restrict__ in,
                                                  _Float16* __restrict__ out, int n) {
    int i = blockIdx.x * 256 + threadIdx.x;
    if (i < n) out[i] = (_Float16)in[i];
}

// Phase 2: recurrent scan. 4 blocks x 256 threads; block handles 16 batch rows.
// h = tanh(h @ W^T + ux_t) with W (f16) + h (f16) resident in LDS.
__global__ __launch_bounds__(256) void rnn_scan(
    const float* __restrict__ W, const float* __restrict__ ux,
    _Float16* __restrict__ hs, float* __restrict__ hlast) {
    __shared__ __align__(16) _Float16 Wsh[Hsz * Hsz];   // 32 KB, W[j][k] row-major
    __shared__ __align__(16) _Float16 hbuf[16 * Hsz];   // 4 KB, h[m][k]

    int tid = threadIdx.x;
    for (int i = tid; i < Hsz * Hsz; i += 256) Wsh[i] = (_Float16)W[i];
    for (int i = tid; i < 16 * Hsz; i += 256) hbuf[i] = (_Float16)0.f;
    __syncthreads();

    const int wave = tid >> 5;
    const int lane = tid & 31;
    const int l = lane & 15;
    const int hi = lane >> 4;              // 0 or 1
    const int nbase = wave * 16;           // each wave owns one 16-col tile of H
    const int b0 = blockIdx.x * 16;
    const int n = nbase + l;

    for (int t = 0; t < Tsz; ++t) {
        // Issue the 8 ux loads first so they are in flight under the WMMAs.
        const float* uxr = ux + ((size_t)t * Bsz + b0) * Hsz;
        float uxv[8];
#pragma unroll
        for (int r = 0; r < 8; ++r) uxv[r] = uxr[(r + hi * 8) * Hsz + n];

        v8f acc = {0.f, 0.f, 0.f, 0.f, 0.f, 0.f, 0.f, 0.f};
#pragma unroll
        for (int kt = 0; kt < 4; ++kt) {
            int k0 = kt * 32;
            const _Float16* ap = hbuf + l * Hsz + k0 + hi * 8;
            v16h a = ld_frag(ap, ap + 16);
            const _Float16* bp = Wsh + (nbase + l) * Hsz + k0 + hi * 8;
            v16h b = ld_frag(bp, bp + 16);
            acc = wmma_f16(a, b, acc);
        }
        __syncthreads();                   // all waves done reading hbuf
#pragma unroll
        for (int r = 0; r < 8; ++r) {
            int m = r + hi * 8;
            float v = fast_tanh(acc[r] + uxv[r]);
            hbuf[m * Hsz + n] = (_Float16)v;
            hs[((size_t)(b0 + m) * Tsz + t) * Hsz + n] = (_Float16)v;
            if (t == Tsz - 1) hlast[(b0 + m) * Hsz + n] = v;
        }
        __syncthreads();                   // new h visible before next step
    }
}

// Phase 3: out[m][v] = hs[m,:] . V_w[v,:] + V_b[v], m = b*T+t.
// One wave per block; wave tile 32(M) x 64(N); K = 128 in 4 WMMA steps.
// A (4 MB) and B (2 MB) are L2-resident; output stores are the HBM bottleneck.
__global__ __launch_bounds__(32) void out_gemm(
    const _Float16* __restrict__ hs, const _Float16* __restrict__ vw,
    const float* __restrict__ vb, float* __restrict__ out) {
    const int lane = threadIdx.x;
    const int l = lane & 15;
    const int hi = lane >> 4;
    const int mbase = blockIdx.x * 32;
    const int nbase = blockIdx.y * 64;

    v8f acc[2][4];
#pragma unroll
    for (int mt = 0; mt < 2; ++mt)
#pragma unroll
        for (int nt = 0; nt < 4; ++nt)
            acc[mt][nt] = (v8f){0.f, 0.f, 0.f, 0.f, 0.f, 0.f, 0.f, 0.f};

#pragma unroll
    for (int kt = 0; kt < 4; ++kt) {
        int k0 = kt * 32;
        v16h a[2], b[4];
#pragma unroll
        for (int mt = 0; mt < 2; ++mt) {
            const _Float16* ap = hs + (size_t)(mbase + mt * 16 + l) * Hsz + k0 + hi * 8;
            a[mt] = ld_frag(ap, ap + 16);
        }
#pragma unroll
        for (int nt = 0; nt < 4; ++nt) {
            const _Float16* bp = vw + (size_t)(nbase + nt * 16 + l) * Hsz + k0 + hi * 8;
            b[nt] = ld_frag(bp, bp + 16);
        }
#pragma unroll
        for (int mt = 0; mt < 2; ++mt)
#pragma unroll
            for (int nt = 0; nt < 4; ++nt)
                acc[mt][nt] = wmma_f16(a[mt], b[nt], acc[mt][nt]);
    }

#pragma unroll
    for (int mt = 0; mt < 2; ++mt) {
#pragma unroll
        for (int nt = 0; nt < 4; ++nt) {
            int n = nbase + nt * 16 + l;
            float bias = vb[n];
#pragma unroll
            for (int r = 0; r < 8; ++r) {
                int m = mbase + mt * 16 + r + hi * 8;
                __builtin_nontemporal_store(acc[mt][nt][r] + bias,
                                            &out[(size_t)m * Vsz + n]);
            }
        }
    }
}

extern "C" void kernel_launch(void* const* d_in, const int* in_sizes, int n_in,
                              void* d_out, int out_size, void* d_ws, size_t ws_size,
                              hipStream_t stream) {
    const float* x  = (const float*)d_in[0];
    const float* Ww = (const float*)d_in[1];
    const float* Wb = (const float*)d_in[2];
    const float* Uw = (const float*)d_in[3];
    const float* Ub = (const float*)d_in[4];
    const float* Vw = (const float*)d_in[5];
    const float* Vb = (const float*)d_in[6];
    float* out = (float*)d_out;
    float* hlast = out + (size_t)Bsz * Tsz * Vsz;

    float*    ux  = (float*)d_ws;                                           // 8 MB
    _Float16* hsb = (_Float16*)((char*)d_ws + (size_t)Tsz * Bsz * Hsz * 4); // 4 MB
    _Float16* vwh = hsb + (size_t)Bsz * Tsz * Hsz;                          // 2 MB

    ux_kernel<<<(Tsz * Bsz * Hsz) / 256, 256, 0, stream>>>(x, Uw, Ub, Wb, ux);
    cvt_kernel<<<(Vsz * Hsz + 255) / 256, 256, 0, stream>>>(Vw, vwh, Vsz * Hsz);
    rnn_scan<<<Bsz / 16, 256, 0, stream>>>(Ww, ux, hsb, hlast);
    dim3 g3(Tsz * Bsz / 32, Vsz / 64);
    out_gemm<<<g3, 32, 0, stream>>>(hsb, vwh, Vb, out);
}
